// DeformableSpatialAttention_86681029968259
// MI455X (gfx1250) — matrix operand order repaired and checked by reading, not compiled, not run on hardware
//
#include <hip/hip_runtime.h>
#include <hip/hip_bf16.h>
#include <cstdint>
#include <cstddef>

typedef __attribute__((ext_vector_type(16))) _Float16 v16h;
typedef __attribute__((ext_vector_type(8)))  _Float16 v8h;
typedef __attribute__((ext_vector_type(2)))  _Float16 v2h;
typedef __attribute__((ext_vector_type(8)))  float    v8f;

static constexpr int kB  = 8;
static constexpr int kC  = 256;
static constexpr int kH  = 128;
static constexpr int kW  = 128;
static constexpr int kHW = kH * kW;      // 16384
static constexpr int kNH = 8;
static constexpr int kHD = 32;

// ---------------------------------------------------------------------------
// Kernel 0b: convert v_weight / o_weight (fp32 [o][c]) to f16 [o][c]
// ---------------------------------------------------------------------------
__global__ void k_wcvt(const float* __restrict__ vw, const float* __restrict__ ow,
                       _Float16* __restrict__ vwh, _Float16* __restrict__ owh) {
    int i = blockIdx.x * 256 + threadIdx.x;
    if (i < kC * kC) {
        vwh[i] = (_Float16)vw[i];
    } else {
        int j = i - kC * kC;
        owh[j] = (_Float16)ow[j];
    }
}

// ---------------------------------------------------------------------------
// Kernel 0a: transpose x (B,C,HW) fp32 -> x_t (B,HW,C) f16, LDS 64x64 tiles
// ---------------------------------------------------------------------------
__global__ void k_xt(const float* __restrict__ x, _Float16* __restrict__ xt) {
    __shared__ float t[64][65];
    int beta = blockIdx.x;
    int b    = beta >> 10;          // 1024 tiles per batch (256 p-tiles * 4 c-tiles)
    int rem  = beta & 1023;
    int pt   = rem >> 2, ct = rem & 3;
    int p0   = pt * 64, c0 = ct * 64;
    int tid  = threadIdx.x;
#pragma unroll
    for (int i = 0; i < 16; ++i) {
        int idx  = tid + i * 256;
        int crow = idx >> 6, pcol = idx & 63;
        t[crow][pcol] = x[((size_t)(b * kC + c0 + crow)) * kHW + p0 + pcol];
    }
    __syncthreads();
#pragma unroll
    for (int i = 0; i < 16; ++i) {
        int idx  = tid + i * 256;
        int prow = idx >> 6, ccol = idx & 63;
        xt[((size_t)(b) * kHW + p0 + prow) * kC + c0 + ccol] = (_Float16)t[ccol][prow];
    }
}

// ---------------------------------------------------------------------------
// Kernel 1: fused depthwise 3x3 (SAME, zero pad) + offset projection + tanh*0.5
// Block = 256 threads = 2 rows of 128 pixels.  Output dyx (B,NH,2,HW) fp32.
// ---------------------------------------------------------------------------
__global__ void k_dwoff(const float* __restrict__ x, const float* __restrict__ dww,
                        const float* __restrict__ offw, const float* __restrict__ offb,
                        float* __restrict__ dyx) {
    __shared__ float swo[16][256];   // off_weight [o][c]
    __shared__ float sx[4][132];     // rows r0-1..r0+2, cols -1..128 (+pad)
    int tid = threadIdx.x;
    int gp0 = blockIdx.x * 256;
    int b   = gp0 >> 14;
    int p0  = gp0 & (kHW - 1);
    int r0  = p0 >> 7;

    for (int i = tid; i < 16 * 256; i += 256) swo[i >> 8][i & 255] = offw[i];

    float off[16];
#pragma unroll
    for (int o = 0; o < 16; ++o) off[o] = 0.f;

    int lr = tid >> 7, col = tid & 127;

    for (int c = 0; c < kC; ++c) {
        __syncthreads();   // protects sx (and swo on first iter)
        for (int idx = tid; idx < 4 * 130; idx += 256) {
            int i  = idx / 130;
            int cc = idx % 130 - 1;
            int gr = r0 - 1 + i;
            float v = 0.f;
            if (gr >= 0 && gr < kH && cc >= 0 && cc < kW)
                v = x[((size_t)(b * kC + c)) * kHW + gr * kW + cc];
            sx[i][idx % 130] = v;
        }
        __syncthreads();
        const float* w9 = dww + c * 9;
        float h = 0.f;
#pragma unroll
        for (int dr = 0; dr < 3; ++dr)
#pragma unroll
            for (int dc = 0; dc < 3; ++dc)
                h = fmaf(w9[dr * 3 + dc], sx[lr + dr][col + dc], h);
#pragma unroll
        for (int o = 0; o < 16; ++o) off[o] = fmaf(h, swo[o][c], off[o]);
    }

    int prow = r0 + lr;
#pragma unroll
    for (int o = 0; o < 16; ++o) {
        float v  = tanhf(off[o] + offb[o]) * 0.5f;
        int head = o >> 1, wh = o & 1;
        dyx[((size_t)((b * kNH + head) * 2 + wh)) * kHW + prow * kW + col] = v;
    }
}

// ---------------------------------------------------------------------------
// Kernel 2: V = x @ v_weight^T via WMMA f16 -> V (B,NH,HW,HD) f16
// Block: 256 thr, tile M=64 pixels x N=256; wave = 2 Mtiles x 4 Ntiles.
// ---------------------------------------------------------------------------
__global__ void __launch_bounds__(256, 1)
k_vgemm(const _Float16* __restrict__ xt, const _Float16* __restrict__ vwh,
        _Float16* __restrict__ V) {
    int beta = blockIdx.x;
    int b    = beta >> 8;
    int p0   = (beta & 255) * 64;
    int tid  = threadIdx.x;
    int wave = tid >> 5, lane = tid & 31;
    int mwave = wave >> 2, nwave = wave & 3;
    int mb = p0 + mwave * 32;
    int nb = nwave * 64;
    int lm = lane & 15;
    int hi = (lane >> 4) & 1;

    v8f acc[2][4] = {};
    const _Float16* xbase = xt + (size_t)b * kHW * kC;

    for (int k0 = 0; k0 < kC; k0 += 32) {
        v16h afr[2], bfr[4];
#pragma unroll
        for (int mt = 0; mt < 2; ++mt) {
            int p  = mb + 16 * mt + lm;
            int c0 = k0 + hi * 16;
            const _Float16* ap = xbase + (size_t)p * kC + c0;
            v8h lo = *(const v8h*)ap;
            v8h hh = *(const v8h*)(ap + 8);
            v16h a;
#pragma unroll
            for (int e = 0; e < 8; ++e) { a[e] = lo[e]; a[e + 8] = hh[e]; }
            afr[mt] = a;
        }
#pragma unroll
        for (int nt = 0; nt < 4; ++nt) {
            int o  = nb + 16 * nt + lm;
            int c0 = k0 + hi * 16;
            bfr[nt] = *(const v16h*)(vwh + (size_t)o * kC + c0);
        }
#pragma unroll
        for (int mt = 0; mt < 2; ++mt)
#pragma unroll
            for (int nt = 0; nt < 4; ++nt)
                acc[mt][nt] = __builtin_amdgcn_wmma_f32_16x16x32_f16(
                    false, afr[mt], false, bfr[nt], (short)0, acc[mt][nt], false, false);
    }

#pragma unroll
    for (int mt = 0; mt < 2; ++mt) {
#pragma unroll
        for (int nt = 0; nt < 4; ++nt) {
            int n0   = nb + 16 * nt;
            int head = n0 >> 5;
            int hd   = (n0 & 31) + lm;
            _Float16* vp = V + (size_t)(b * kNH + head) * kHW * kHD + hd;
#pragma unroll
            for (int r = 0; r < 8; ++r) {
                int p = mb + 16 * mt + r + hi * 8;
                vp[(size_t)p * kHD] = (_Float16)acc[mt][nt][r];
            }
        }
    }
}

// ---------------------------------------------------------------------------
// Kernel 3: bilinear sampling of V into LDS + O-GEMM via WMMA +
// deterministic per-(b,head) partial sums.  out_pre (B,HW,C) fp32.
// ---------------------------------------------------------------------------
__global__ void __launch_bounds__(256, 1)
k_sample_ogemm(const _Float16* __restrict__ V, const float* __restrict__ dyx,
               const _Float16* __restrict__ owh, float* __restrict__ outp,
               float* __restrict__ partials) {
    __shared__ _Float16 sA[64 * 264];      // 64 pixels x 256 ch (+8 pad halves)
    __shared__ float sPart[8][4][2];
    int beta = blockIdx.x;
    int b    = beta >> 8;
    int p0   = (beta & 255) * 64;
    int tid  = threadIdx.x;

    // ---- phase 1: gather-sample V into LDS tile ----
    for (int t = tid; t < 512; t += 256) {
        int head = t >> 6;
        int pl   = t & 63;
        int p    = p0 + pl;
        int r    = p >> 7, cc = p & 127;
        float dy = dyx[((size_t)((b * kNH + head) * 2 + 0)) * kHW + p];
        float dx = dyx[((size_t)((b * kNH + head) * 2 + 1)) * kHW + p];
        float gy = -1.f + 2.f * (float)r * (1.f / 127.f);
        float gx = -1.f + 2.f * (float)cc * (1.f / 127.f);
        float sy = fminf(fmaxf(gy + dy, -1.f), 1.f);
        float sx = fminf(fmaxf(gx + dx, -1.f), 1.f);
        float iy = (sy + 1.f) * 0.5f * 127.f;
        float ix = (sx + 1.f) * 0.5f * 127.f;
        float y0f = floorf(iy), x0f = floorf(ix);
        float wy = iy - y0f, wx = ix - x0f;
        int y0 = min(max((int)y0f, 0), kH - 1);
        int y1 = min(max((int)y0f + 1, 0), kH - 1);
        int x0 = min(max((int)x0f, 0), kW - 1);
        int x1 = min(max((int)x0f + 1, 0), kW - 1);
        const _Float16* vb = V + (size_t)(b * kNH + head) * kHW * kHD;
        int   q[4]  = { y0 * kW + x0, y0 * kW + x1, y1 * kW + x0, y1 * kW + x1 };
        float wt[4] = { (1.f - wx) * (1.f - wy), wx * (1.f - wy),
                        (1.f - wx) * wy,         wx * wy };
        float acc[32];
#pragma unroll
        for (int e = 0; e < 32; ++e) acc[e] = 0.f;
#pragma unroll
        for (int tp = 0; tp < 4; ++tp) {
            const v8h* vp = (const v8h*)(vb + (size_t)q[tp] * kHD);
            float wgt = wt[tp];
#pragma unroll
            for (int j = 0; j < 4; ++j) {
                v8h hv = vp[j];
#pragma unroll
                for (int e = 0; e < 8; ++e)
                    acc[j * 8 + e] = fmaf(wgt, (float)hv[e], acc[j * 8 + e]);
            }
        }
        _Float16* srow = sA + pl * 264 + head * 32;
#pragma unroll
        for (int j = 0; j < 16; ++j) {
            v2h pk;
            pk[0] = (_Float16)acc[2 * j];
            pk[1] = (_Float16)acc[2 * j + 1];
            *(v2h*)(srow + 2 * j) = pk;
        }
    }
    __syncthreads();

    // ---- phase 2: WMMA GEMM against o_weight ----
    int wave = tid >> 5, lane = tid & 31;
    int mwave = wave >> 2, nwave = wave & 3;
    int mbl = mwave * 32;
    int nb  = nwave * 64;
    int lm  = lane & 15;
    int hi  = (lane >> 4) & 1;

    v8f acc[2][4] = {};
    for (int k0 = 0; k0 < kC; k0 += 32) {
        v16h afr[2], bfr[4];
#pragma unroll
        for (int mt = 0; mt < 2; ++mt) {
            int pl = mbl + 16 * mt + lm;
            int c0 = k0 + hi * 16;
            const _Float16* ap = sA + pl * 264 + c0;
            v8h lo = *(const v8h*)ap;
            v8h hh = *(const v8h*)(ap + 8);
            v16h a;
#pragma unroll
            for (int e = 0; e < 8; ++e) { a[e] = lo[e]; a[e + 8] = hh[e]; }
            afr[mt] = a;
        }
#pragma unroll
        for (int nt = 0; nt < 4; ++nt) {
            int o  = nb + 16 * nt + lm;
            int c0 = k0 + hi * 16;
            bfr[nt] = *(const v16h*)(owh + (size_t)o * kC + c0);
        }
#pragma unroll
        for (int mt = 0; mt < 2; ++mt)
#pragma unroll
            for (int nt = 0; nt < 4; ++nt)
                acc[mt][nt] = __builtin_amdgcn_wmma_f32_16x16x32_f16(
                    false, afr[mt], false, bfr[nt], (short)0, acc[mt][nt], false, false);
    }

    // ---- store out_pre + per-head partial sums (deterministic) ----
    float psum[4] = {0.f, 0.f, 0.f, 0.f};
    float psq[4]  = {0.f, 0.f, 0.f, 0.f};
#pragma unroll
    for (int mt = 0; mt < 2; ++mt) {
#pragma unroll
        for (int nt = 0; nt < 4; ++nt) {
            int o = nb + 16 * nt + lm;
            float s = 0.f, sq = 0.f;
#pragma unroll
            for (int r = 0; r < 8; ++r) {
                float v = acc[mt][nt][r];
                int p = p0 + mbl + 16 * mt + r + hi * 8;
                outp[((size_t)b * kHW + p) * kC + o] = v;
                s += v;
                sq += v * v;
            }
            psum[nt] += s;
            psq[nt]  += sq;
        }
    }
#pragma unroll
    for (int nt = 0; nt < 4; ++nt) {
        float s = psum[nt], sq = psq[nt];
#pragma unroll
        for (int off = 16; off >= 1; off >>= 1) {
            s  += __shfl_xor(s, off, 32);
            sq += __shfl_xor(sq, off, 32);
        }
        if (lane == 0) { sPart[wave][nt][0] = s; sPart[wave][nt][1] = sq; }
    }
    __syncthreads();
    if (tid < 16) {
        int h = tid >> 1, part = tid & 1;
        float tot = 0.f;
        for (int w = 0; w < 8; ++w)
            for (int j = 0; j < 4; ++j) {
                int hw = ((w & 3) * 2) + (j >> 1);
                if (hw == h) tot += sPart[w][j][part];
            }
        partials[(size_t)beta * 16 + h * 2 + part] = tot;
    }
}

// ---------------------------------------------------------------------------
// Kernel 4: reduce 256 block partials per (b,head) -> mean, rstd
// ---------------------------------------------------------------------------
__global__ void k_stats(const float* __restrict__ partials, float* __restrict__ stats) {
    int g = threadIdx.x;
    if (g >= kB * kNH) return;
    int b = g >> 3, h = g & 7;
    float s = 0.f, q = 0.f;
    for (int t = 0; t < 256; ++t) {
        size_t base = ((size_t)(b * 256 + t)) * 16 + h * 2;
        s += partials[base];
        q += partials[base + 1];
    }
    const float invN = 1.f / (float)(kHD * kHW);   // 1/524288
    float mu  = s * invN;
    float var = q * invN - mu * mu;
    stats[g * 2]     = mu;
    stats[g * 2 + 1] = rsqrtf(var + 1e-5f);
}

// ---------------------------------------------------------------------------
// Kernel 5: layernorm affine + residual + transpose (B,HW,C)->(B,C,HW)
// ---------------------------------------------------------------------------
__global__ void k_final(const float* __restrict__ outp, const float* __restrict__ stats,
                        const float* __restrict__ gamma, const float* __restrict__ beta_,
                        const float* __restrict__ x, float* __restrict__ out) {
    __shared__ float t[64][65];
    int bId = blockIdx.x;
    int b   = bId >> 10;
    int rem = bId & 1023;
    int pt  = rem >> 2, ot = rem & 3;
    int p0  = pt * 64, o0 = ot * 64;
    int tid = threadIdx.x;
#pragma unroll
    for (int i = 0; i < 16; ++i) {
        int idx  = tid + i * 256;
        int prow = idx >> 6, ocol = idx & 63;
        int o    = o0 + ocol;
        int head = o >> 5;
        float mu = stats[(b * kNH + head) * 2];
        float rs = stats[(b * kNH + head) * 2 + 1];
        float v  = outp[((size_t)b * kHW + p0 + prow) * kC + o];
        t[prow][ocol] = (v - mu) * rs * gamma[o] + beta_[o];
    }
    __syncthreads();
#pragma unroll
    for (int i = 0; i < 16; ++i) {
        int idx  = tid + i * 256;
        int orow = idx >> 6, pcol = idx & 63;
        size_t a = ((size_t)(b * kC + o0 + orow)) * kHW + p0 + pcol;
        out[a] = x[a] + t[pcol][orow];
    }
}

// ---------------------------------------------------------------------------
extern "C" void kernel_launch(void* const* d_in, const int* in_sizes, int n_in,
                              void* d_out, int out_size, void* d_ws, size_t ws_size,
                              hipStream_t stream) {
    (void)in_sizes; (void)n_in; (void)out_size; (void)ws_size;
    const float* x    = (const float*)d_in[0];
    const float* dww  = (const float*)d_in[1];
    const float* offw = (const float*)d_in[2];
    const float* offb = (const float*)d_in[3];
    const float* vw   = (const float*)d_in[4];
    const float* ow   = (const float*)d_in[5];
    const float* gmm  = (const float*)d_in[6];
    const float* bta  = (const float*)d_in[7];
    float* out = (float*)d_out;

    char* ws = (char*)d_ws;
    _Float16* xt    = (_Float16*)(ws);                    // 67,108,864 B
    _Float16* vwh   = (_Float16*)(ws + 67108864);         //    131,072 B
    _Float16* owh   = (_Float16*)(ws + 67239936);         //    131,072 B
    float*    dyx   = (float*)   (ws + 67371008);         // 16,777,216 B
    _Float16* V     = (_Float16*)(ws + 84148224);         // 67,108,864 B
    float*    outp  = (float*)   (ws + 151257088);        // 134,217,728 B
    float*    part  = (float*)   (ws + 285474816);        //    131,072 B
    float*    stats = (float*)   (ws + 285605888);        //        512 B

    k_wcvt        <<<512,  256, 0, stream>>>(vw, ow, vwh, owh);
    k_xt          <<<8192, 256, 0, stream>>>(x, xt);
    k_dwoff       <<<512,  256, 0, stream>>>(x, dww, offw, offb, dyx);
    k_vgemm       <<<2048, 256, 0, stream>>>(xt, vwh, V);
    k_sample_ogemm<<<2048, 256, 0, stream>>>(V, dyx, owh, outp, part);
    k_stats       <<<1,    64,  0, stream>>>(part, stats);
    k_final       <<<8192, 256, 0, stream>>>(outp, stats, gmm, bta, x, out);
}
